// SCPerformerSelfAttention_39281770889425
// MI455X (gfx1250) — matrix operand order repaired and checked by reading, not compiled, not run on hardware
//
#include <hip/hip_runtime.h>
#include <hip/hip_bf16.h>
#include <math.h>

typedef __bf16 bf16;
typedef __attribute__((ext_vector_type(16))) bf16  v16bf;
typedef __attribute__((ext_vector_type(8)))  float v8f;

#define DEVFN __device__ __forceinline__

constexpr int kB     = 2;
constexpr int kN     = 4096;
constexpr int kH     = 1024;
constexpr int kHeads = 16;
constexpr int kD     = 64;
constexpr int kM     = 256;
constexpr int kRows  = kB * kN;          // 8192 rows for the H-wide GEMMs
constexpr int kBHN   = kB * kHeads * kN; // 131072 flattened (b,h,n) rows
constexpr int TBK    = 32;               // K-step per WMMA (bf16 16x16x32)

// ---------------------------------------------------------------------------
// WMMA helpers (CDNA5 wave32, v_wmma_f32_16x16x32_bf16)
// ---------------------------------------------------------------------------
DEVFN v8f wmma_bf16(v16bf a, v16bf b, v8f c) {
  // 8 args: (neg_a, A, neg_b, B, c_mod, C, reuse_a, reuse_b)
  return __builtin_amdgcn_wmma_f32_16x16x32_bf16(false, a, false, b, (short)0,
                                                 c, false, false);
}

// A fragment: tile stored row-major [16][TBK] (k contiguous).
// Lane L (0..15): M=L, K = 0..7 (VGPR0-3) and 16..23 (VGPR4-7).
// Lane L+16:      M=L, K = 8..15 and 24..31.  -> two 16B contiguous reads.
DEVFN v16bf frag_a(const bf16* tile, int lane) {
  const int m  = lane & 15;
  const int kb = (lane >> 4) << 3; // 0 or 8
  const bf16* p = tile + m * TBK + kb;
  union { uint4 u[2]; v16bf v; } x;
  x.u[0] = *reinterpret_cast<const uint4*>(p);
  x.u[1] = *reinterpret_cast<const uint4*>(p + 16);
  return x.v;
}

// B fragment: tile stored transposed [16 n][TBK k] (k contiguous).
// Lane L (0..15): N=L, K=0..7 and 16..23; lane L+16: N=L, K=8..15 and 24..31.
DEVFN v16bf frag_b(const bf16* tile, int lane) {
  const int n  = lane & 15;
  const int kb = (lane >> 4) << 3; // 0 or 8
  const bf16* p = tile + n * TBK + kb;
  union { uint4 u[2]; v16bf v; } x;
  x.u[0] = *reinterpret_cast<const uint4*>(p);
  x.u[1] = *reinterpret_cast<const uint4*>(p + 16);
  return x.v;
}

// ---------------------------------------------------------------------------
// CDNA5 async global->LDS copy (GLOBAL_LOAD_ASYNC_TO_LDS_B128, ASYNCcnt).
// Per-lane: 16 bytes from gsrc -> LDS at lds_dst.  Low 32 bits of the generic
// shared-pointer are the LDS byte offset (flat LDS aperture encoding).
// ---------------------------------------------------------------------------
DEVFN void async_ld_b128(void* lds_dst, const void* gsrc) {
  uint32_t loff  = (uint32_t)(uintptr_t)lds_dst;
  uint64_t gaddr = (uint64_t)(uintptr_t)gsrc;
  asm volatile("global_load_async_to_lds_b128 %0, %1, off"
               :
               : "v"(loff), "v"(gaddr)
               : "memory");
}
DEVFN void wait_async0() {
  asm volatile("s_wait_asynccnt 0" ::: "memory");
}

// ---------------------------------------------------------------------------
// fp32 -> bf16 conversion (optionally scaled)
// ---------------------------------------------------------------------------
__global__ void cvt_f32_bf16(const float* __restrict__ src,
                             bf16* __restrict__ dst, int n, float scale) {
  int i = blockIdx.x * 256 + threadIdx.x;
  if (i < n) dst[i] = (bf16)(src[i] * scale);
}

// ---------------------------------------------------------------------------
// QKV projection: C = X @ W^T + b, X[8192,1024] bf16, W[1024,1024] (row = out
// feature, so the B-transposed tile is just W rows). Output in [b,h,n,d] bf16.
// Block tile 128(M) x 64(N); 8 waves in a 4x2 grid, each wave owns a 32x32
// output (4 accumulators, 2x fragment reuse).  Staging via async-to-LDS.
// grid = (64, 16, 3); block = 256.
// ---------------------------------------------------------------------------
__global__ __launch_bounds__(256) void qkv_gemm(
    const bf16* __restrict__ X, const bf16* __restrict__ Wqb,
    const bf16* __restrict__ Wkb, const bf16* __restrict__ Wvb,
    const float* __restrict__ bq, const float* __restrict__ bk,
    const float* __restrict__ bv, bf16* __restrict__ Qd,
    bf16* __restrict__ Kd, bf16* __restrict__ Vd) {
  const bf16* W = Wqb; const float* bias = bq; bf16* dst = Qd;
  if (blockIdx.z == 1) { W = Wkb; bias = bk; dst = Kd; }
  else if (blockIdx.z == 2) { W = Wvb; bias = bv; dst = Vd; }

  __shared__ alignas(16) bf16 As[128 * TBK];
  __shared__ alignas(16) bf16 Bs[64 * TBK];

  const int gm   = blockIdx.x * 128;
  const int gn   = blockIdx.y * 64;
  const int lane = threadIdx.x & 31;
  const int wave = threadIdx.x >> 5;
  const int wm   = (wave >> 1) * 32;
  const int wn   = (wave & 1) * 32;

  v8f z = {0.f, 0.f, 0.f, 0.f, 0.f, 0.f, 0.f, 0.f};
  v8f acc00 = z, acc01 = z, acc10 = z, acc11 = z;

  const int t  = threadIdx.x;
  const int ar = t >> 1;              // 0..127 (A row)
  const int ac = (t & 1) << 4;        // 0 or 16 (A k-col)
  const int br = t >> 2;              // 0..63  (B row = out feature)
  const int bc = (t & 3) << 3;        // 0..24  (B k-col)

  for (int k0 = 0; k0 < kH; k0 += TBK) {
    async_ld_b128(&As[ar * TBK + ac],     &X[(size_t)(gm + ar) * kH + k0 + ac]);
    async_ld_b128(&As[ar * TBK + ac + 8], &X[(size_t)(gm + ar) * kH + k0 + ac + 8]);
    async_ld_b128(&Bs[br * TBK + bc],     &W[(size_t)(gn + br) * kH + k0 + bc]);
    wait_async0();
    __syncthreads();
    v16bf a0 = frag_a(&As[wm * TBK], lane);
    v16bf a1 = frag_a(&As[(wm + 16) * TBK], lane);
    v16bf b0 = frag_b(&Bs[wn * TBK], lane);
    v16bf b1 = frag_b(&Bs[(wn + 16) * TBK], lane);
    acc00 = wmma_bf16(a0, b0, acc00);
    acc01 = wmma_bf16(a0, b1, acc01);
    acc10 = wmma_bf16(a1, b0, acc10);
    acc11 = wmma_bf16(a1, b1, acc11);
    __syncthreads();
  }

  // C/D layout: col = lane&15; rows = r + 8*(lane>>4), r = 0..7
  const int colL = lane & 15;
  const int rh   = (lane >> 4) << 3;
  const int oA = gn + wn + colL;
  const int oB = oA + 16;
  const float biA = bias[oA];
  const float biB = bias[oB];
  const int hA = oA >> 6, dA = oA & 63;
  const int hB = oB >> 6, dB = oB & 63;
#pragma unroll
  for (int r = 0; r < 8; ++r) {
    const int row0 = gm + wm + rh + r;
    const int row1 = row0 + 16;
    const int b0i = row0 >> 12, n0 = row0 & (kN - 1);
    const int b1i = row1 >> 12, n1 = row1 & (kN - 1);
    dst[(((size_t)b0i * kHeads + hA) * kN + n0) * kD + dA] = (bf16)(acc00[r] + biA);
    dst[(((size_t)b0i * kHeads + hB) * kN + n0) * kD + dB] = (bf16)(acc01[r] + biB);
    dst[(((size_t)b1i * kHeads + hA) * kN + n1) * kD + dA] = (bf16)(acc10[r] + biA);
    dst[(((size_t)b1i * kHeads + hB) * kN + n1) * kD + dB] = (bf16)(acc11[r] + biB);
  }
}

// ---------------------------------------------------------------------------
// diag_k[row] = -0.5 * sum_d k[row,d]^2  and per-block max partials.
// grid = 256 blocks x 512 threads (one thread per (b,h,n) row).
// ---------------------------------------------------------------------------
__global__ void diag_max_kernel(const bf16* __restrict__ Kd,
                                float* __restrict__ diag,
                                float* __restrict__ pmax) {
  const int row = blockIdx.x * 512 + threadIdx.x;
  const uint4* p4 = reinterpret_cast<const uint4*>(Kd + (size_t)row * kD);
  float s = 0.f;
#pragma unroll
  for (int i = 0; i < 8; ++i) {
    union { uint4 u; bf16 h8[8]; } t;
    t.u = p4[i];
#pragma unroll
    for (int j = 0; j < 8; ++j) { float x = (float)t.h8[j]; s += x * x; }
  }
  const float dval = -0.5f * s;
  diag[row] = dval;
  __shared__ float red[512];
  red[threadIdx.x] = dval;
  __syncthreads();
  for (int off = 256; off > 0; off >>= 1) {
    if ((int)threadIdx.x < off)
      red[threadIdx.x] = fmaxf(red[threadIdx.x], red[threadIdx.x + off]);
    __syncthreads();
  }
  if (threadIdx.x == 0) pmax[blockIdx.x] = red[0];
}

__global__ void reduce_max_kernel(const float* __restrict__ pmax,
                                  float* __restrict__ stab) {
  __shared__ float red[256];
  red[threadIdx.x] = pmax[threadIdx.x];
  __syncthreads();
  for (int off = 128; off > 0; off >>= 1) {
    if ((int)threadIdx.x < off)
      red[threadIdx.x] = fmaxf(red[threadIdx.x], red[threadIdx.x + off]);
    __syncthreads();
  }
  if (threadIdx.x == 0) stab[0] = red[0];
}

// ---------------------------------------------------------------------------
// Feature map: data_dash = data @ (normalizer*proj)^T, then
//   query: qf = M^-0.5 * (exp(data_dash) + eps)          (stabilizer cancels)
//   key:   kf = M^-0.5 * (exp(diag + data_dash - stab) + eps)
// A rows = flattened (b,h,n), lda = 64; B-transposed tile = projb rows.
// grid = (2048, 4); block = 256.
// ---------------------------------------------------------------------------
__global__ __launch_bounds__(256) void feat_gemm(
    const bf16* __restrict__ Data, const bf16* __restrict__ Pb,
    const float* __restrict__ diag, const float* __restrict__ stab,
    bf16* __restrict__ Out, int is_query) {
  __shared__ alignas(16) bf16 As[64 * TBK];
  __shared__ alignas(16) bf16 Bs[64 * TBK];
  const int gm   = blockIdx.x * 64;  // flat (b,h,n)
  const int gn   = blockIdx.y * 64;  // feature m
  const int lane = threadIdx.x & 31;
  const int wave = threadIdx.x >> 5;
  const int wm   = (wave >> 1) * 16;
  const int wn   = (wave & 1) * 32;
  v8f acc0 = {0.f, 0.f, 0.f, 0.f, 0.f, 0.f, 0.f, 0.f};
  v8f acc1 = acc0;
  const int lr = threadIdx.x >> 2;
  const int lc = (threadIdx.x & 3) << 3;
  for (int k0 = 0; k0 < kD; k0 += TBK) {
    *reinterpret_cast<uint4*>(&As[lr * TBK + lc]) =
        *reinterpret_cast<const uint4*>(&Data[(size_t)(gm + lr) * kD + k0 + lc]);
    *reinterpret_cast<uint4*>(&Bs[lr * TBK + lc]) =
        *reinterpret_cast<const uint4*>(&Pb[(size_t)(gn + lr) * kD + k0 + lc]);
    __syncthreads();
    v16bf a  = frag_a(&As[wm * TBK], lane);
    v16bf b0 = frag_b(&Bs[wn * TBK], lane);
    v16bf b1 = frag_b(&Bs[(wn + 16) * TBK], lane);
    acc0 = wmma_bf16(a, b0, acc0);
    acc1 = wmma_bf16(a, b1, acc1);
    __syncthreads();
  }
  const int colL = lane & 15;
  const int rh   = (lane >> 4) << 3;
  const float stabv = is_query ? 0.f : stab[0];
  float dg[8];
#pragma unroll
  for (int r = 0; r < 8; ++r)
    dg[r] = is_query ? 0.f : diag[gm + wm + rh + r];
  const int o0 = gn + wn + colL;
#pragma unroll
  for (int r = 0; r < 8; ++r) {
    const int row = gm + wm + rh + r;
    const float v0 = 0.0625f * (__expf(dg[r] + acc0[r] - stabv) + 1e-3f);
    const float v1 = 0.0625f * (__expf(dg[r] + acc1[r] - stabv) + 1e-3f);
    Out[(size_t)row * kM + o0]      = (bf16)v0;
    Out[(size_t)row * kM + o0 + 16] = (bf16)v1;
  }
}

// k_cumsum[bh,m] = sum_n kf[bh,n,m].  grid = 32 blocks x 256 threads.
__global__ void ksum_kernel(const bf16* __restrict__ kf,
                            float* __restrict__ ksum) {
  const int bh = blockIdx.x;
  const int m  = threadIdx.x;
  const bf16* p = kf + (size_t)bh * kN * kM + m;
  float s = 0.f;
  for (int n = 0; n < kN; ++n) s += (float)p[(size_t)n * kM];
  ksum[bh * kM + m] = s;
}

// d_inv[row] = 1 / dot(qf[row,:], ksum[bh,:]).  grid = 512 x 256.
__global__ void dinv_kernel(const bf16* __restrict__ qf,
                            const float* __restrict__ ksum,
                            float* __restrict__ dinv) {
  const int row = blockIdx.x * 256 + threadIdx.x;
  const int bh  = row >> 12;
  const bf16*  p  = qf + (size_t)row * kM;
  const float* ks = ksum + bh * kM;
  float s = 0.f;
  for (int m = 0; m < kM; ++m) s += (float)p[m] * ks[m];
  dinv[row] = 1.f / s;
}

// ---------------------------------------------------------------------------
// context[bh,m,d] = sum_n kf[bh,n,m] * v[bh,n,d].  A = kf^T (LDS transpose),
// B^T = v^T (LDS transpose).  grid = (4 m-chunks, 32 bh); K = 4096.
// ---------------------------------------------------------------------------
__global__ __launch_bounds__(256) void ctx_gemm(const bf16* __restrict__ kf,
                                                const bf16* __restrict__ Vd,
                                                bf16* __restrict__ ctx) {
  __shared__ alignas(16) bf16 As[64 * TBK]; // As[m][n_k]
  __shared__ alignas(16) bf16 Bs[64 * TBK]; // Bs[d][n_k]
  const int gm   = blockIdx.x * 64;
  const int bh   = blockIdx.y;
  const int lane = threadIdx.x & 31;
  const int wave = threadIdx.x >> 5;
  const int wm   = (wave >> 1) * 16;
  const int wn   = (wave & 1) * 32;
  v8f acc0 = {0.f, 0.f, 0.f, 0.f, 0.f, 0.f, 0.f, 0.f};
  v8f acc1 = acc0;
  const int tn = threadIdx.x >> 3;        // 0..31 : k (=n) row
  const int tc = (threadIdx.x & 7) << 3;  // 0..56 : contiguous cols
  for (int k0 = 0; k0 < kN; k0 += TBK) {
    union { uint4 u; bf16 h8[8]; } ta, tb;
    ta.u = *reinterpret_cast<const uint4*>(
        &kf[((size_t)bh * kN + k0 + tn) * kM + gm + tc]);
    tb.u = *reinterpret_cast<const uint4*>(
        &Vd[((size_t)bh * kN + k0 + tn) * kD + tc]);
#pragma unroll
    for (int j = 0; j < 8; ++j) {
      As[(tc + j) * TBK + tn] = ta.h8[j];
      Bs[(tc + j) * TBK + tn] = tb.h8[j];
    }
    __syncthreads();
    v16bf a  = frag_a(&As[wm * TBK], lane);
    v16bf b0 = frag_b(&Bs[wn * TBK], lane);
    v16bf b1 = frag_b(&Bs[(wn + 16) * TBK], lane);
    acc0 = wmma_bf16(a, b0, acc0);
    acc1 = wmma_bf16(a, b1, acc1);
    __syncthreads();
  }
  const int colL = lane & 15;
  const int rh   = (lane >> 4) << 3;
  const int d0   = wn + colL;
#pragma unroll
  for (int r = 0; r < 8; ++r) {
    const int m = gm + wm + rh + r;
    ctx[((size_t)bh * kM + m) * kD + d0]      = (bf16)acc0[r];
    ctx[((size_t)bh * kM + m) * kD + d0 + 16] = (bf16)acc1[r];
  }
}

// ---------------------------------------------------------------------------
// out[bh,n,d] = (qf[bh,n,:] @ ctx[bh,:,d]) * d_inv[bh,n], written merged as
// attn[b][n][h*64+d] bf16.  grid = (64 n-chunks, 32 bh); K = 256.
// ---------------------------------------------------------------------------
__global__ __launch_bounds__(256) void out_gemm(const bf16* __restrict__ qf,
                                                const bf16* __restrict__ ctx,
                                                const float* __restrict__ dinv,
                                                bf16* __restrict__ attn) {
  __shared__ alignas(16) bf16 As[64 * TBK]; // As[n][m_k]
  __shared__ alignas(16) bf16 Bs[64 * TBK]; // Bs[d][m_k]
  const int gm   = blockIdx.x * 64;  // n offset within bh
  const int bh   = blockIdx.y;
  const int b    = bh >> 4;
  const int h    = bh & 15;
  const int lane = threadIdx.x & 31;
  const int wave = threadIdx.x >> 5;
  const int wm   = (wave >> 1) * 16;
  const int wn   = (wave & 1) * 32;
  v8f acc0 = {0.f, 0.f, 0.f, 0.f, 0.f, 0.f, 0.f, 0.f};
  v8f acc1 = acc0;
  const int lr = threadIdx.x >> 2;        // 0..63 : qf rows
  const int lc = (threadIdx.x & 3) << 3;  // 0..24
  const int tn = threadIdx.x >> 3;        // 0..31 : ctx m rows
  const int tc = (threadIdx.x & 7) << 3;  // 0..56 : d cols
  for (int k0 = 0; k0 < kM; k0 += TBK) {
    *reinterpret_cast<uint4*>(&As[lr * TBK + lc]) =
        *reinterpret_cast<const uint4*>(
            &qf[((size_t)bh * kN + gm + lr) * kM + k0 + lc]);
    union { uint4 u; bf16 h8[8]; } tb;
    tb.u = *reinterpret_cast<const uint4*>(
        &ctx[((size_t)bh * kM + k0 + tn) * kD + tc]);
#pragma unroll
    for (int j = 0; j < 8; ++j) Bs[(tc + j) * TBK + tn] = tb.h8[j];
    __syncthreads();
    v16bf a  = frag_a(&As[wm * TBK], lane);
    v16bf b0 = frag_b(&Bs[wn * TBK], lane);
    v16bf b1 = frag_b(&Bs[(wn + 16) * TBK], lane);
    acc0 = wmma_bf16(a, b0, acc0);
    acc1 = wmma_bf16(a, b1, acc1);
    __syncthreads();
  }
  const int colL = lane & 15;
  const int rh   = (lane >> 4) << 3;
  const int d0   = wn + colL;
#pragma unroll
  for (int r = 0; r < 8; ++r) {
    const int n = gm + wm + rh + r;
    const float di = dinv[(size_t)bh * kN + n];
    attn[((size_t)b * kN + n) * kH + h * kD + d0]      = (bf16)(acc0[r] * di);
    attn[((size_t)b * kN + n) * kH + h * kD + d0 + 16] = (bf16)(acc1[r] * di);
  }
}

// ---------------------------------------------------------------------------
// Final: out = attn @ Wo^T + bo, fp32 output.  128x64 block tile, async
// staging, 4 WMMAs per wave per K-step.  grid = (64, 16).
// ---------------------------------------------------------------------------
__global__ __launch_bounds__(256) void final_gemm(const bf16* __restrict__ X,
                                                  const bf16* __restrict__ W,
                                                  const float* __restrict__ bo,
                                                  float* __restrict__ out) {
  __shared__ alignas(16) bf16 As[128 * TBK];
  __shared__ alignas(16) bf16 Bs[64 * TBK];
  const int gm   = blockIdx.x * 128;
  const int gn   = blockIdx.y * 64;
  const int lane = threadIdx.x & 31;
  const int wave = threadIdx.x >> 5;
  const int wm   = (wave >> 1) * 32;
  const int wn   = (wave & 1) * 32;
  v8f z = {0.f, 0.f, 0.f, 0.f, 0.f, 0.f, 0.f, 0.f};
  v8f acc00 = z, acc01 = z, acc10 = z, acc11 = z;
  const int t  = threadIdx.x;
  const int ar = t >> 1;
  const int ac = (t & 1) << 4;
  const int br = t >> 2;
  const int bc = (t & 3) << 3;
  for (int k0 = 0; k0 < kH; k0 += TBK) {
    async_ld_b128(&As[ar * TBK + ac],     &X[(size_t)(gm + ar) * kH + k0 + ac]);
    async_ld_b128(&As[ar * TBK + ac + 8], &X[(size_t)(gm + ar) * kH + k0 + ac + 8]);
    async_ld_b128(&Bs[br * TBK + bc],     &W[(size_t)(gn + br) * kH + k0 + bc]);
    wait_async0();
    __syncthreads();
    v16bf a0 = frag_a(&As[wm * TBK], lane);
    v16bf a1 = frag_a(&As[(wm + 16) * TBK], lane);
    v16bf b0 = frag_b(&Bs[wn * TBK], lane);
    v16bf b1 = frag_b(&Bs[(wn + 16) * TBK], lane);
    acc00 = wmma_bf16(a0, b0, acc00);
    acc01 = wmma_bf16(a0, b1, acc01);
    acc10 = wmma_bf16(a1, b0, acc10);
    acc11 = wmma_bf16(a1, b1, acc11);
    __syncthreads();
  }
  const int colL = lane & 15;
  const int rh   = (lane >> 4) << 3;
  const int oA = gn + wn + colL;
  const int oB = oA + 16;
  const float biA = bo[oA];
  const float biB = bo[oB];
#pragma unroll
  for (int r = 0; r < 8; ++r) {
    const int row0 = gm + wm + rh + r;
    const int row1 = row0 + 16;
    out[(size_t)row0 * kH + oA] = acc00[r] + biA;
    out[(size_t)row0 * kH + oB] = acc01[r] + biB;
    out[(size_t)row1 * kH + oA] = acc10[r] + biA;
    out[(size_t)row1 * kH + oB] = acc11[r] + biB;
  }
}

// ---------------------------------------------------------------------------
extern "C" void kernel_launch(void* const* d_in, const int* in_sizes, int n_in,
                              void* d_out, int out_size, void* d_ws,
                              size_t ws_size, hipStream_t stream) {
  const float* hidden = (const float*)d_in[0];
  const float* Wq = (const float*)d_in[1];
  const float* bq = (const float*)d_in[2];
  const float* Wk = (const float*)d_in[3];
  const float* bk = (const float*)d_in[4];
  const float* Wv = (const float*)d_in[5];
  const float* bv = (const float*)d_in[6];
  const float* Wo = (const float*)d_in[7];
  const float* bo = (const float*)d_in[8];
  const float* proj = (const float*)d_in[9];
  float* out = (float*)d_out;

  char* ws = (char*)d_ws;
  auto carve = [&](size_t bytes) -> char* {
    char* p = ws;
    ws += (bytes + 255) & ~(size_t)255;
    return p;
  };
  bf16* hb    = (bf16*)carve((size_t)kRows * kH * 2);
  bf16* wqb   = (bf16*)carve((size_t)kH * kH * 2);
  bf16* wkb   = (bf16*)carve((size_t)kH * kH * 2);
  bf16* wvb   = (bf16*)carve((size_t)kH * kH * 2);
  bf16* wob   = (bf16*)carve((size_t)kH * kH * 2);
  bf16* projb = (bf16*)carve((size_t)kM * kD * 2);
  bf16* qb    = (bf16*)carve((size_t)kBHN * kD * 2);
  bf16* kb    = (bf16*)carve((size_t)kBHN * kD * 2);
  bf16* vb    = (bf16*)carve((size_t)kBHN * kD * 2);
  bf16* qf    = (bf16*)carve((size_t)kBHN * kM * 2);
  bf16* kf    = (bf16*)carve((size_t)kBHN * kM * 2);
  float* diag = (float*)carve((size_t)kBHN * 4);
  float* pmax = (float*)carve(256 * 4);
  float* stab = (float*)carve(256);
  float* ksum = (float*)carve((size_t)kB * kHeads * kM * 4);
  float* dinv = (float*)carve((size_t)kBHN * 4);
  bf16* ctx   = (bf16*)carve((size_t)kB * kHeads * kM * kD * 2);
  bf16* attn  = (bf16*)carve((size_t)kRows * kH * 2);

  // Stage 0: conversions (proj pre-scaled by DHEAD^-0.25)
  cvt_f32_bf16<<<(kRows * kH + 255) / 256, 256, 0, stream>>>(hidden, hb,
                                                             kRows * kH, 1.f);
  cvt_f32_bf16<<<(kH * kH + 255) / 256, 256, 0, stream>>>(Wq, wqb, kH * kH, 1.f);
  cvt_f32_bf16<<<(kH * kH + 255) / 256, 256, 0, stream>>>(Wk, wkb, kH * kH, 1.f);
  cvt_f32_bf16<<<(kH * kH + 255) / 256, 256, 0, stream>>>(Wv, wvb, kH * kH, 1.f);
  cvt_f32_bf16<<<(kH * kH + 255) / 256, 256, 0, stream>>>(Wo, wob, kH * kH, 1.f);
  cvt_f32_bf16<<<(kM * kD + 255) / 256, 256, 0, stream>>>(
      proj, projb, kM * kD, 0.35355339059327379f);

  // Stage 1: Q/K/V projections (WMMA, async-to-LDS staging)
  qkv_gemm<<<dim3(kRows / 128, kH / 64, 3), 256, 0, stream>>>(
      hb, wqb, wkb, wvb, bq, bk, bv, qb, kb, vb);

  // Stage 2: key diag + global stabilizer
  diag_max_kernel<<<kBHN / 512, 512, 0, stream>>>(kb, diag, pmax);
  reduce_max_kernel<<<1, 256, 0, stream>>>(pmax, stab);

  // Stage 3: feature maps (WMMA + exp epilogue)
  feat_gemm<<<dim3(kBHN / 64, kM / 64), 256, 0, stream>>>(qb, projb, diag,
                                                          stab, qf, 1);
  feat_gemm<<<dim3(kBHN / 64, kM / 64), 256, 0, stream>>>(kb, projb, diag,
                                                          stab, kf, 0);

  // Stage 4: normalizers
  ksum_kernel<<<kB * kHeads, kM, 0, stream>>>(kf, ksum);
  dinv_kernel<<<kBHN / 256, 256, 0, stream>>>(qf, ksum, dinv);

  // Stage 5: linear attention (WMMA)
  ctx_gemm<<<dim3(kM / 64, kB * kHeads), 256, 0, stream>>>(kf, vb, ctx);
  out_gemm<<<dim3(kN / 64, kB * kHeads), 256, 0, stream>>>(qf, ctx, dinv, attn);

  // Stage 6: output projection (WMMA, fp32 out + bias, async staging)
  final_gemm<<<dim3(kRows / 128, kH / 64), 256, 0, stream>>>(attn, wob, bo, out);

  (void)in_sizes; (void)n_in; (void)out_size; (void)ws_size;
}
